// DocREModel_50611894616328
// MI455X (gfx1250) — matrix operand order repaired and checked by reading, not compiled, not run on hardware
//
#include <hip/hip_runtime.h>
#include <hip/hip_fp16.h>

typedef __attribute__((ext_vector_type(16))) _Float16 v16h;
typedef __attribute__((ext_vector_type(8)))  _Float16 h8;
typedef __attribute__((ext_vector_type(8)))  float    v8f;

// ---------------- problem constants ----------------
constexpr int B_ = 4, C_ = 1024, H_ = 768, NH_ = 12;
constexpr int NE_ = 24, NL_ = 97, EMB_ = 1024;
constexpr int P_ = NE_ * NE_;            // 576
constexpr int N_ = B_ * P_;              // 2304
constexpr int NE_R_ = 25;
constexpr int TWO_H = 2 * H_;            // 1536
constexpr int KBIL = EMB_ * 64;          // 65536
constexpr int NLP = 112;                 // NL padded to 16
constexpr int CIN = 2 * NL_ + 1;         // 195
constexpr int CINP = 224;                // K padded to mult 32
constexpr int L_ = 20;
constexpr int RL_ = NL_ * L_;            // 1940
constexpr int RLP = 1952;                // padded to mult 32
constexpr int MN_ = NE_R_ * NE_R_;       // 625
constexpr int MNP = 640;                 // padded to mult 32

// ---------------- WMMA fragment layout (wave32) ----------------
// A (16x32 f16): lane&15 -> M; slots 0-7 = K (k0 + 8*(lane>=16) + s),
//                slots 8-15 = K (+16).  => two contiguous 16B chunks.
// B (32x16 f16): lane&15 -> N, slot s -> K as for A.  => 16 strided loads (stride ldb).
// C/D (f32): lane&15 -> N, elem r -> M = r + 8*(lane>=16).

__device__ __forceinline__ v16h wmma_a_load(const _Float16* p) {
    h8 lo = *(const h8*)(p);
    h8 hi = *(const h8*)(p + 16);
    return __builtin_shufflevector(lo, hi, 0,1,2,3,4,5,6,7,8,9,10,11,12,13,14,15);
}

// ---------------- small prep kernels ----------------
__global__ void k_ent_emb(const float* __restrict__ seq, const int* __restrict__ mpos,
                          float* __restrict__ ent) {
    int blk = blockIdx.x;                       // B*NE
    int b = blk / NE_;
    int p0 = mpos[blk * 3 + 0] + 1, p1 = mpos[blk * 3 + 1] + 1, p2 = mpos[blk * 3 + 2] + 1;
    const float* s0 = seq + ((long long)b * C_ + p0) * H_;
    const float* s1 = seq + ((long long)b * C_ + p1) * H_;
    const float* s2 = seq + ((long long)b * C_ + p2) * H_;
    for (int d = threadIdx.x; d < H_; d += blockDim.x) {
        float a = s0[d], c = s1[d], e = s2[d];
        float mx = fmaxf(a, fmaxf(c, e));
        ent[(long long)blk * H_ + d] = mx + __logf(__expf(a - mx) + __expf(c - mx) + __expf(e - mx));
    }
}

__global__ void k_ent_att(const float* __restrict__ att, const int* __restrict__ mpos,
                          float* __restrict__ eatt) {
    int blk = blockIdx.x;                       // B*NE*NH
    int nh = blk % NH_;
    int be = blk / NH_;
    int b  = be / NE_;
    int p0 = mpos[be * 3 + 0] + 1, p1 = mpos[be * 3 + 1] + 1, p2 = mpos[be * 3 + 2] + 1;
    const float* base = att + ((long long)b * NH_ + nh) * C_ * C_;
    for (int c = threadIdx.x; c < C_; c += blockDim.x) {
        float v = base[(long long)p0 * C_ + c] + base[(long long)p1 * C_ + c] + base[(long long)p2 * C_ + c];
        eatt[((long long)be * NH_ + nh) * C_ + c] = v * (1.f / 3.f);
    }
}

__global__ void k_htatt(const float* __restrict__ eatt, const int* __restrict__ hts,
                        _Float16* __restrict__ out) {
    int blk = blockIdx.x;                       // B*P
    int b = blk / P_;
    int h = hts[blk * 2 + 0], t = hts[blk * 2 + 1];
    const float* hA = eatt + ((long long)(b * NE_ + h) * NH_) * C_;
    const float* tA = eatt + ((long long)(b * NE_ + t) * NH_) * C_;
    float v[4];
    float mysum = 0.f;
#pragma unroll
    for (int q = 0; q < 4; ++q) {
        int cc = (q << 8) + threadIdx.x;
        float acc = 0.f;
        for (int nh = 0; nh < NH_; ++nh)
            acc += hA[nh * C_ + cc] * tA[nh * C_ + cc];
        v[q] = acc * (1.f / 12.f);
        mysum += v[q];
    }
    __shared__ float red[256];
    red[threadIdx.x] = mysum;
    __syncthreads();
    for (int s = 128; s > 0; s >>= 1) {
        if ((int)threadIdx.x < s) red[threadIdx.x] += red[threadIdx.x + s];
        __syncthreads();
    }
    float inv = 1.f / (red[0] + 1e-5f);
#pragma unroll
    for (int q = 0; q < 4; ++q) {
        int cc = (q << 8) + threadIdx.x;
        out[(long long)blk * C_ + cc] = (_Float16)(v[q] * inv);
    }
}

__global__ void k_f32_to_f16(const float* __restrict__ src, _Float16* __restrict__ dst, long long n) {
    long long i = (long long)blockIdx.x * blockDim.x + threadIdx.x;
    if (i < n) dst[i] = (_Float16)src[i];
}

__global__ void k_wbil_pad(const float* __restrict__ src, _Float16* __restrict__ dst) {
    long long i = (long long)blockIdx.x * blockDim.x + threadIdx.x;
    if (i >= (long long)KBIL * NLP) return;
    long long k = i / NLP;
    int n = (int)(i % NLP);
    dst[i] = (n < NL_) ? (_Float16)src[k * NL_ + n] : (_Float16)0.f;
}

__global__ void k_zero(float* __restrict__ p, long long n) {
    long long i = (long long)blockIdx.x * blockDim.x + threadIdx.x;
    if (i < n) p[i] = 0.f;
}
__global__ void k_zero_h(_Float16* __restrict__ p, long long n) {
    long long i = (long long)blockIdx.x * blockDim.x + threadIdx.x;
    if (i < n) p[i] = (_Float16)0.f;
}

__global__ void k_pack(const float* __restrict__ ent, const float* __restrict__ rs,
                       const int* __restrict__ hts,
                       _Float16* __restrict__ Xh, _Float16* __restrict__ Xt) {
    int np = blockIdx.x;                        // N
    int b = np / P_;
    int h = hts[np * 2 + 0], t = hts[np * 2 + 1];
    const float* eh = ent + (long long)(b * NE_ + h) * H_;
    const float* et = ent + (long long)(b * NE_ + t) * H_;
    const float* rr = rs + (long long)np * H_;
    for (int d = threadIdx.x; d < TWO_H; d += blockDim.x) {
        float rv = (d < H_) ? 0.f : rr[d - H_];
        float hv = (d < H_) ? eh[d] : rv;
        float tv = (d < H_) ? et[d] : rv;
        Xh[(long long)np * TWO_H + d] = (_Float16)hv;
        Xt[(long long)np * TWO_H + d] = (_Float16)tv;
    }
}

// ---------------- generic batched WMMA GEMM (padded, guard-free) ----------------
// Requires: M%32==0, N%32==0, K%32==0; buffers sized to padded dims.
// 32x32 output per wave (2x2 WMMA; A and B fragments each reused twice),
// 8 waves/block, grid.y = batch.
__global__ void __launch_bounds__(256)
k_gemm(const _Float16* __restrict__ Ag, const _Float16* __restrict__ Bg,
       float* __restrict__ Cf, _Float16* __restrict__ Ch,
       const float* __restrict__ bias, int do_tanh,
       int M, int Nn, int K, int lda, int ldb, int ldc,
       long long sA, long long sB, long long sC) {
    const _Float16* A = Ag + (long long)blockIdx.y * sA;
    const _Float16* B = Bg + (long long)blockIdx.y * sB;
    int lane = threadIdx.x & 31;
    int wave = threadIdx.x >> 5;
    int Mt2 = M >> 5, Nt2 = Nn >> 5;
    int tile = blockIdx.x * 8 + wave;
    if (tile >= Mt2 * Nt2) return;
    int mt = tile / Nt2, nt = tile % Nt2;
    int m0 = mt << 5, n0 = nt << 5;
    int mr = lane & 15, rh = lane >> 4;
    const _Float16* arow0 = A + (long long)(m0 + mr) * lda + (rh << 3);
    const _Float16* arow1 = arow0 + (long long)16 * lda;
    v8f c00 = {}, c01 = {}, c10 = {}, c11 = {};
    for (int k0 = 0; k0 < K; k0 += 32) {
        v16h a0 = wmma_a_load(arow0 + k0);
        v16h a1 = wmma_a_load(arow1 + k0);
        const _Float16* brow = B + (long long)(k0 + (rh << 3)) * ldb + n0 + mr;
        v16h b0, b1;
        for (int s = 0; s < 16; ++s) {
            long long kk = (long long)((s & 7) + ((s >> 3) << 4)) * ldb;
            b0[s] = brow[kk];
            b1[s] = brow[kk + 16];
        }
        c00 = __builtin_amdgcn_wmma_f32_16x16x32_f16(false, a0, false, b0, (short)0, c00, false, false);
        c01 = __builtin_amdgcn_wmma_f32_16x16x32_f16(false, a0, false, b1, (short)0, c01, false, false);
        c10 = __builtin_amdgcn_wmma_f32_16x16x32_f16(false, a1, false, b0, (short)0, c10, false, false);
        c11 = __builtin_amdgcn_wmma_f32_16x16x32_f16(false, a1, false, b1, (short)0, c11, false, false);
    }
    int ns0 = n0 + mr, ns1 = ns0 + 16;
    float bs0 = bias ? bias[ns0] : 0.f;
    float bs1 = bias ? bias[ns1] : 0.f;
#pragma unroll
    for (int r = 0; r < 8; ++r) {
        int ms0 = m0 + r + (rh << 3);
        int ms1 = ms0 + 16;
        long long base0 = (long long)blockIdx.y * sC + (long long)ms0 * ldc;
        long long base1 = (long long)blockIdx.y * sC + (long long)ms1 * ldc;
        float v00 = c00[r] + bs0, v01 = c01[r] + bs1;
        float v10 = c10[r] + bs0, v11 = c11[r] + bs1;
        if (do_tanh) { v00 = tanhf(v00); v01 = tanhf(v01); v10 = tanhf(v10); v11 = tanhf(v11); }
        if (Ch) {
            Ch[base0 + ns0] = (_Float16)v00; Ch[base0 + ns1] = (_Float16)v01;
            Ch[base1 + ns0] = (_Float16)v10; Ch[base1 + ns1] = (_Float16)v11;
        } else {
            Cf[base0 + ns0] = v00; Cf[base0 + ns1] = v01;
            Cf[base1 + ns0] = v10; Cf[base1 + ns1] = v11;
        }
    }
}

// ---------------- bilinear logits: on-the-fly A = hs[m,gi] * ts[m,gj] ----------------
// 32 rows per block (two 16-row m-tiles per wave share each W_bil B fragment).
__global__ void __launch_bounds__(256)
k_bilinear(const _Float16* __restrict__ hsp, const _Float16* __restrict__ tsp,
           const _Float16* __restrict__ Wb, const float* __restrict__ bias,
           float* __restrict__ logits) {
    __shared__ _Float16 sh[32 * EMB_];
    __shared__ _Float16 st[32 * EMB_];
    int r0 = blockIdx.x << 5;
    const uint32_t* hsrc = (const uint32_t*)(hsp + (long long)r0 * EMB_);
    const uint32_t* tsrc = (const uint32_t*)(tsp + (long long)r0 * EMB_);
    uint32_t* hd = (uint32_t*)sh;
    uint32_t* td = (uint32_t*)st;
    for (int i = threadIdx.x; i < 32 * EMB_ / 2; i += 256) { hd[i] = hsrc[i]; td[i] = tsrc[i]; }
    __syncthreads();
    int wave = threadIdx.x >> 5;
    if (wave >= 7) return;                       // 7 col tiles cover 112 >= 97
    int lane = threadIdx.x & 31;
    int mr = lane & 15, rh = lane >> 4;
    int n0 = wave << 4;
    const _Float16* hrow0 = sh + mr * EMB_;
    const _Float16* hrow1 = sh + (16 + mr) * EMB_;
    const _Float16* trow0 = st + mr * EMB_ + (rh << 3);
    const _Float16* trow1 = st + (16 + mr) * EMB_ + (rh << 3);
    v8f c0 = {}, c1 = {};
    for (int g = 0; g < 16; ++g) {
        for (int i = 0; i < 64; ++i) {
            _Float16 hv0 = hrow0[(g << 6) + i];
            _Float16 hv1 = hrow1[(g << 6) + i];
            int kbase = ((g << 6) + i) << 6;     // *64
#pragma unroll
            for (int jh = 0; jh < 2; ++jh) {
                int tb = (g << 6) + (jh << 5);
                h8 t0lo = *(const h8*)(trow0 + tb);
                h8 t0hi = *(const h8*)(trow0 + tb + 16);
                h8 t1lo = *(const h8*)(trow1 + tb);
                h8 t1hi = *(const h8*)(trow1 + tb + 16);
                h8 a0lo = t0lo * hv0, a0hi = t0hi * hv0;
                h8 a1lo = t1lo * hv1, a1hi = t1hi * hv1;
                v16h a0 = __builtin_shufflevector(a0lo, a0hi, 0,1,2,3,4,5,6,7,8,9,10,11,12,13,14,15);
                v16h a1 = __builtin_shufflevector(a1lo, a1hi, 0,1,2,3,4,5,6,7,8,9,10,11,12,13,14,15);
                const _Float16* brow = Wb + (long long)(kbase + (jh << 5) + (rh << 3)) * NLP + n0 + mr;
                v16h b;
                for (int s = 0; s < 16; ++s)
                    b[s] = brow[(long long)((s & 7) + ((s >> 3) << 4)) * NLP];
                c0 = __builtin_amdgcn_wmma_f32_16x16x32_f16(false, a0, false, b, (short)0, c0, false, false);
                c1 = __builtin_amdgcn_wmma_f32_16x16x32_f16(false, a1, false, b, (short)0, c1, false, false);
            }
        }
    }
    int ns = n0 + mr;
    if (ns < NL_) {
        float bs = bias[ns];
#pragma unroll
        for (int r = 0; r < 8; ++r) {
            int ms0 = r0 + r + (rh << 3);
            int ms1 = ms0 + 16;
            logits[(long long)ms0 * NL_ + ns] = c0[r] + bs;
            logits[(long long)ms1 * NL_ + ns] = c1[r] + bs;
        }
    }
}

// ---------------- softmax over logits -> dense scatter ----------------
__global__ void k_softmax_scatter(const float* __restrict__ logits, const int* __restrict__ hts,
                                  float* __restrict__ dense) {
    int blk = blockIdx.x;                        // B*P
    int b = blk / P_;
    int h = hts[blk * 2 + 0], t = hts[blk * 2 + 1];
    int tid = threadIdx.x;                       // 128
    float v = (tid < NL_) ? logits[(long long)blk * NL_ + tid] : -3.0e38f;
    __shared__ float red[128];
    red[tid] = v; __syncthreads();
    for (int s = 64; s > 0; s >>= 1) { if (tid < s) red[tid] = fmaxf(red[tid], red[tid + s]); __syncthreads(); }
    float mx = red[0]; __syncthreads();
    float e = (tid < NL_) ? __expf(v - mx) : 0.f;
    red[tid] = e; __syncthreads();
    for (int s = 64; s > 0; s >>= 1) { if (tid < s) red[tid] += red[tid + s]; __syncthreads(); }
    float inv = 1.f / red[0];
    if (tid < NL_)
        dense[(((long long)b * NE_R_ + h) * NE_R_ + t) * NL_ + tid] = e * inv;
}

// ---------------- masked softmax on diff_w -> w0T, w1 ; dwt ----------------
__global__ void k_prep_w(const float* __restrict__ diff_w, const float* __restrict__ diff_weights,
                         _Float16* __restrict__ w0T, _Float16* __restrict__ w1,
                         float* __restrict__ dwt) {
    int blk = blockIdx.x;                        // RL_
    int r = blk / L_, l = blk % L_;
    int tid = threadIdx.x;                       // 256
    __shared__ float red[256];
    for (int t = 0; t < 2; ++t) {
        float v = -3.0e38f;
        if (tid < CIN) {
            v = diff_w[(((long long)r * 2 + t) * L_ + l) * CIN + tid];
            if (r > 0 && (tid == 0 || tid == NL_)) v += -1e30f;
        }
        red[tid] = v; __syncthreads();
        for (int s = 128; s > 0; s >>= 1) { if (tid < s) red[tid] = fmaxf(red[tid], red[tid + s]); __syncthreads(); }
        float mx = red[0]; __syncthreads();
        float e = (tid < CIN) ? __expf(v - mx) : 0.f;
        red[tid] = e; __syncthreads();
        for (int s = 128; s > 0; s >>= 1) { if (tid < s) red[tid] += red[tid + s]; __syncthreads(); }
        float inv = 1.f / red[0]; __syncthreads();
        if (tid < CINP) {
            _Float16 hv = (tid < CIN) ? (_Float16)(e * inv) : (_Float16)0.f;
            if (t == 0) w0T[(long long)tid * RLP + blk] = hv;   // [CINP, RLP]
            else        w1[(long long)blk * CINP + tid] = hv;   // [RLP, CINP]
        }
    }
    if (tid == 0) dwt[blk] = tanhf(diff_weights[blk]);
}

// ---------------- build inp / inpT (padded rows/cols written as zero) ----------------
__global__ void k_prep_inp(const float* __restrict__ dense, _Float16* __restrict__ inp,
                           _Float16* __restrict__ inpT) {
    int blk = blockIdx.x;                        // B*MNP
    int b = blk / MNP, mn = blk % MNP;
    int c = threadIdx.x;
    if (c >= CINP) return;
    float v = 0.f;
    if (mn < MN_) {
        int m = mn / NE_R_, n = mn % NE_R_;
        if (c < NL_)           v = dense[(((long long)b * NE_R_ + m) * NE_R_ + n) * NL_ + c];
        else if (c < 2 * NL_)  v = dense[(((long long)b * NE_R_ + n) * NE_R_ + m) * NL_ + (c - NL_)];
        else if (c == 2 * NL_) v = (m == n) ? 1.f : 0.f;
    }
    _Float16 hv = (_Float16)v;
    inp[((long long)b * MNP + mn) * CINP + c] = hv;
    inpT[((long long)b * CINP + c) * MNP + mn] = hv;
}

// ---------------- s1 + readout ----------------
// rules[b,m,n,r] = sum_l dwt[r,l] * sum_k v[r,l,m,k] * s0[(k*25+n), r*20+l]
__global__ void k_rules(const float* __restrict__ Vmat, const float* __restrict__ S0,
                        const float* __restrict__ dwt, float* __restrict__ rules) {
    long long idx = (long long)blockIdx.x * blockDim.x + threadIdx.x;
    if (idx >= (long long)B_ * MN_ * NL_) return;
    int r  = (int)(idx % NL_);
    int mn = (int)((idx / NL_) % MN_);
    int b  = (int)(idx / ((long long)NL_ * MN_));
    int m = mn / NE_R_, n = mn % NE_R_;
    const float* Vb = Vmat + (long long)b * RLP * MNP;
    const float* Sb = S0   + (long long)b * MNP * RLP;
    float acc = 0.f;
    for (int l = 0; l < L_; ++l) {
        int rl = r * L_ + l;
        const float* vrow = Vb + (long long)rl * MNP + m * NE_R_;
        const float* srow = Sb + (long long)n * RLP + rl;
        float t = 0.f;
#pragma unroll 5
        for (int k = 0; k < NE_R_; ++k)
            t += vrow[k] * srow[(long long)k * NE_R_ * RLP];
        acc += dwt[rl] * t;
    }
    rules[idx] = acc;
}

// ---------------- final gather + add ----------------
__global__ void k_final(const float* __restrict__ logits, const float* __restrict__ rules,
                        const int* __restrict__ hts, float* __restrict__ out) {
    long long idx = (long long)blockIdx.x * blockDim.x + threadIdx.x;
    if (idx >= (long long)N_ * NL_) return;
    int r = (int)(idx % NL_);
    int np = (int)(idx / NL_);
    int b = np / P_;
    int h = hts[np * 2 + 0], t = hts[np * 2 + 1];
    out[idx] = logits[idx] + rules[((long long)b * MN_ + h * NE_R_ + t) * NL_ + r];
}

// ---------------- host side ----------------
extern "C" void kernel_launch(void* const* d_in, const int* in_sizes, int n_in,
                              void* d_out, int out_size, void* d_ws, size_t ws_size,
                              hipStream_t stream) {
    (void)in_sizes; (void)n_in; (void)out_size; (void)ws_size;
    const float* seq     = (const float*)d_in[0];
    const float* att     = (const float*)d_in[1];
    const int*   mpos    = (const int*)d_in[2];
    const int*   hts     = (const int*)d_in[3];
    const float* W_head  = (const float*)d_in[4];
    const float* b_head  = (const float*)d_in[5];
    const float* W_tail  = (const float*)d_in[6];
    const float* b_tail  = (const float*)d_in[7];
    const float* W_bil   = (const float*)d_in[8];
    const float* b_bil   = (const float*)d_in[9];
    const float* diff_w  = (const float*)d_in[10];
    const float* diff_wt = (const float*)d_in[11];
    float* out = (float*)d_out;

    char* ws = (char*)d_ws;
    size_t off = 0;
    auto alloc = [&](size_t bytes) -> void* {
        void* p = (void*)(ws + off);
        off += (bytes + 255) & ~(size_t)255;
        return p;
    };

    float*    ent_emb = (float*)   alloc((size_t)B_ * NE_ * H_ * 4);
    float*    ent_att = (float*)   alloc((size_t)B_ * NE_ * NH_ * C_ * 4);
    _Float16* htatt16 = (_Float16*)alloc((size_t)N_ * C_ * 2);
    _Float16* seq16   = (_Float16*)alloc((size_t)B_ * C_ * H_ * 2);
    float*    rs      = (float*)   alloc((size_t)N_ * H_ * 4);
    _Float16* Xh      = (_Float16*)alloc((size_t)N_ * TWO_H * 2);
    _Float16* Xt      = (_Float16*)alloc((size_t)N_ * TWO_H * 2);
    _Float16* Wh16    = (_Float16*)alloc((size_t)TWO_H * EMB_ * 2);
    _Float16* Wt16    = (_Float16*)alloc((size_t)TWO_H * EMB_ * 2);
    _Float16* Wb16    = (_Float16*)alloc((size_t)KBIL * NLP * 2);
    _Float16* hs_p    = (_Float16*)alloc((size_t)N_ * EMB_ * 2);
    _Float16* ts_p    = (_Float16*)alloc((size_t)N_ * EMB_ * 2);
    float*    logits  = (float*)   alloc((size_t)N_ * NL_ * 4);
    float*    dense   = (float*)   alloc((size_t)B_ * MN_ * NL_ * 4);
    _Float16* inp     = (_Float16*)alloc((size_t)B_ * MNP * CINP * 2);
    _Float16* inpT    = (_Float16*)alloc((size_t)B_ * CINP * MNP * 2);
    _Float16* w0T     = (_Float16*)alloc((size_t)CINP * RLP * 2);
    _Float16* w1      = (_Float16*)alloc((size_t)RLP * CINP * 2);
    float*    dwt     = (float*)   alloc((size_t)RL_ * 4);
    float*    S0      = (float*)   alloc((size_t)B_ * MNP * RLP * 4);
    float*    Vmat    = (float*)   alloc((size_t)B_ * RLP * MNP * 4);
    float*    rules   = (float*)   alloc((size_t)B_ * MN_ * NL_ * 4);

    // 1. entity embeddings + attentions
    k_ent_emb<<<B_ * NE_, 256, 0, stream>>>(seq, mpos, ent_emb);
    k_ent_att<<<B_ * NE_ * NH_, 256, 0, stream>>>(att, mpos, ent_att);
    // 2. pair attention weights (f16)
    k_htatt<<<N_, 256, 0, stream>>>(ent_att, hts, htatt16);
    // 3. conversions
    {
        long long n = (long long)B_ * C_ * H_;
        k_f32_to_f16<<<(int)((n + 255) / 256), 256, 0, stream>>>(seq, seq16, n);
        n = (long long)TWO_H * EMB_;
        k_f32_to_f16<<<(int)((n + 255) / 256), 256, 0, stream>>>(W_head, Wh16, n);
        k_f32_to_f16<<<(int)((n + 255) / 256), 256, 0, stream>>>(W_tail, Wt16, n);
        n = (long long)KBIL * NLP;
        k_wbil_pad<<<(int)((n + 255) / 256), 256, 0, stream>>>(W_bil, Wb16);
    }
    // 4. rs = ht_att @ seq  (batched)  [576x1024]x[1024x768]
    {
        int tiles = (P_ >> 5) * (H_ >> 5);
        dim3 g((tiles + 7) / 8, B_);
        k_gemm<<<g, 256, 0, stream>>>(htatt16, seq16, rs, nullptr, nullptr, 0,
                                      P_, H_, C_, C_, H_, H_,
                                      (long long)P_ * C_, (long long)C_ * H_, (long long)P_ * H_);
    }
    // 5. pack [hs|rs], [ts|rs] as f16
    k_pack<<<N_, 256, 0, stream>>>(ent_emb, rs, hts, Xh, Xt);
    // 6. head/tail projections with tanh -> f16
    {
        int tiles = (N_ >> 5) * (EMB_ >> 5);
        dim3 g((tiles + 7) / 8, 1);
        k_gemm<<<g, 256, 0, stream>>>(Xh, Wh16, nullptr, hs_p, b_head, 1,
                                      N_, EMB_, TWO_H, TWO_H, EMB_, EMB_, 0, 0, 0);
        k_gemm<<<g, 256, 0, stream>>>(Xt, Wt16, nullptr, ts_p, b_tail, 1,
                                      N_, EMB_, TWO_H, TWO_H, EMB_, EMB_, 0, 0, 0);
    }
    // 7. bilinear logits (K = 65536 synthesized on the fly), 32 rows/block
    k_bilinear<<<N_ / 32, 256, 0, stream>>>(hs_p, ts_p, Wb16, b_bil, logits);
    // 8. softmax -> dense pair matrix
    {
        long long n = (long long)B_ * MN_ * NL_;
        k_zero<<<(int)((n + 255) / 256), 256, 0, stream>>>(dense, n);
        k_softmax_scatter<<<N_, 128, 0, stream>>>(logits, hts, dense);
    }
    // 9. diffusion weights + inputs (zero pad regions first)
    {
        long long n = (long long)CINP * RLP;
        k_zero_h<<<(int)((n + 255) / 256), 256, 0, stream>>>(w0T, n);
        n = (long long)RLP * CINP;
        k_zero_h<<<(int)((n + 255) / 256), 256, 0, stream>>>(w1, n);
    }
    k_prep_w<<<RL_, 256, 0, stream>>>(diff_w, diff_wt, w0T, w1, dwt);
    k_prep_inp<<<B_ * MNP, 256, 0, stream>>>(dense, inp, inpT);
    // 10. s0 = inp @ w0T   [640x224]x[224x1952], batched
    {
        int tiles = (MNP >> 5) * (RLP >> 5);
        dim3 g((tiles + 7) / 8, B_);
        k_gemm<<<g, 256, 0, stream>>>(inp, w0T, S0, nullptr, nullptr, 0,
                                      MNP, RLP, CINP, CINP, RLP, RLP,
                                      (long long)MNP * CINP, 0, (long long)MNP * RLP);
    }
    // 11. v = w1 @ inpT    [1952x224]x[224x640], batched (A shared)
    {
        int tiles = (RLP >> 5) * (MNP >> 5);
        dim3 g((tiles + 7) / 8, B_);
        k_gemm<<<g, 256, 0, stream>>>(w1, inpT, Vmat, nullptr, nullptr, 0,
                                      RLP, MNP, CINP, CINP, MNP, MNP,
                                      0, (long long)CINP * MNP, (long long)RLP * MNP);
    }
    // 12. s1 contraction + readout (tiny, VALU)
    {
        long long n = (long long)B_ * MN_ * NL_;
        k_rules<<<(int)((n + 127) / 128), 128, 0, stream>>>(Vmat, S0, dwt, rules);
    }
    // 13. final: logits_rule + logits
    {
        long long n = (long long)N_ * NL_;
        k_final<<<(int)((n + 255) / 256), 256, 0, stream>>>(logits, rules, hts, out);
    }
}
